// RelPositionMultiHeadedAttention_77352361001223
// MI455X (gfx1250) — matrix operand order repaired and checked by reading, not compiled, not run on hardware
//
#include <hip/hip_runtime.h>
#include <cstddef>

// ---------------------------------------------------------------------------
// RelPositionMultiHeadedAttention for MI455X (gfx1250, wave32, WMMA bf16)
// B=16, T=S=1024, n_feat=512, H=8, D=64.
// Compute-limited (~86 GFLOP vs ~10us of HBM time at 23.3 TB/s) -> bf16 WMMA
// (v_wmma_f32_16x16x32_bf16) + ASYNCcnt-pipelined global->LDS DMA staging
// (global_load_async_to_lds_b128, s_wait_asynccnt).
// ---------------------------------------------------------------------------

typedef __bf16 bf16_t;
typedef __attribute__((ext_vector_type(16))) __bf16        bf16x16;
typedef __attribute__((ext_vector_type(8)))  float         f32x8;
typedef __attribute__((ext_vector_type(4)))  float         f32x4;
typedef __attribute__((ext_vector_type(4)))  unsigned int  u32x4;

#define NB  16
#define NT  1024
#define NF  512
#define NH  8
#define ND  64

__device__ __forceinline__ bf16x16 ld_frag(const bf16_t* p0, const bf16_t* p1) {
  union { bf16x16 v; u32x4 q[2]; } u;
  u.q[0] = *(const u32x4*)p0;   // 16 bytes (8 bf16)
  u.q[1] = *(const u32x4*)p1;   // 16 bytes (8 bf16)
  return u.v;
}

__device__ __forceinline__ f32x8 wmma_bf16(bf16x16 a, bf16x16 b, f32x8 c) {
  // (neg_a, A, neg_b, B, c_mod, C, reuse_a, reuse_b)
  return __builtin_amdgcn_wmma_f32_16x16x32_bf16(false, a, false, b, (short)0, c,
                                                 false, false);
}

__device__ __forceinline__ f32x8 zero8() {
  f32x8 z = {0.f, 0.f, 0.f, 0.f, 0.f, 0.f, 0.f, 0.f};
  return z;
}

// Generic LDS pointer -> LDS byte address (aperture scheme: addr[31:0]).
__device__ __forceinline__ unsigned lds_off(const void* p) {
  return (unsigned)(size_t)p;
}

// Async DMA: 16B per lane, global -> LDS, tracked by ASYNCcnt.
__device__ __forceinline__ void async_copy_b128(unsigned lds_addr, const void* src) {
  asm volatile("global_load_async_to_lds_b128 %0, %1, off"
               :
               : "v"(lds_addr), "v"((unsigned long long)(size_t)src)
               : "memory");
}
__device__ __forceinline__ void wait_async() {
  asm volatile("s_wait_asynccnt 0x0" ::: "memory");
}

// ---------------------------------------------------------------------------
// Prep 1: fp32 -> bf16, 8 elements/thread (b128 store).
// ---------------------------------------------------------------------------
__global__ __launch_bounds__(256) void cvt_bf16(const float* __restrict__ in,
                                                bf16_t* __restrict__ out, int n) {
  int i = (blockIdx.x * 256 + threadIdx.x) * 8;
  if (i + 8 <= n) {
    f32x4 a0 = *(const f32x4*)&in[i];
    f32x4 a1 = *(const f32x4*)&in[i + 4];
    union { bf16_t h[8]; u32x4 q; } pk;
#pragma unroll
    for (int j = 0; j < 4; ++j) {
      pk.h[j]     = (bf16_t)a0[j];
      pk.h[4 + j] = (bf16_t)a1[j];
    }
    *(u32x4*)&out[i] = pk.q;
  }
}

// ---------------------------------------------------------------------------
// Prep 2: W (K,N) fp32 -> WT (N,K) bf16, 32x32 LDS-tiled (coalesced both ways).
// ---------------------------------------------------------------------------
__global__ __launch_bounds__(256) void wtrans_bf16(const float* __restrict__ W,
                                                   bf16_t* __restrict__ WT,
                                                   int K, int N) {
  __shared__ float tile[32][33];
  const int kb = blockIdx.y * 32, nb = blockIdx.x * 32;
  const int tx = threadIdx.x & 31, ty = threadIdx.x >> 5;  // ty 0..7
#pragma unroll
  for (int i = 0; i < 32; i += 8)
    tile[ty + i][tx] = W[(size_t)(kb + ty + i) * N + nb + tx];
  __syncthreads();
#pragma unroll
  for (int i = 0; i < 32; i += 8)
    WT[(size_t)(nb + ty + i) * K + kb + tx] = (bf16_t)tile[tx][ty + i];
}

// ---------------------------------------------------------------------------
// GEMM: C[M,N] = A[M,K] @ W + bias, with A bf16 row-major and WT = W^T bf16
// (N,K) row-major so B-fragments (32x16) are contiguous 2x16B per-lane loads.
// Block tile 128x128, BK=32, 8 waves, wave tile 32x64.
// Double-buffered LDS, staged entirely by async-to-LDS DMA:
//   iter i: s_wait_asynccnt 0 (own tile-i DMA) -> barrier (all waves' data
//   in LDS) -> issue tile i+1 into other buffer -> 8 WMMAs on tile i.
// The single barrier is sufficient: tile i+1 issue is post-barrier, i.e.
// after every wave finished reading that buffer in iteration i-1.
// ---------------------------------------------------------------------------
template <bool OUT_F32>
__global__ __launch_bounds__(256) void gemm_bf16tt(
    const bf16_t* __restrict__ A,   // (M,K) bf16
    const bf16_t* __restrict__ WT,  // (N,K) bf16
    const float* __restrict__ bias, void* __restrict__ Cout,
    int M, int N, int K)
{
  __shared__ bf16_t ldsA[2][128][32];
  __shared__ bf16_t ldsW[2][128][32];

  const int tid   = threadIdx.x;
  const int lane  = tid & 31;
  const int wid   = tid >> 5;
  const int laneR = lane & 15;
  const int laneH = lane >> 4;
  const int bm = blockIdx.y * 128;
  const int bn = blockIdx.x * 128;
  const int wm = (wid & 3) * 32;
  const int wn = (wid >> 2) * 64;

  f32x8 acc[2][4];
#pragma unroll
  for (int mt = 0; mt < 2; ++mt)
#pragma unroll
    for (int nt = 0; nt < 4; ++nt) acc[mt][nt] = zero8();

  // 8KB per tile per matrix = 512 x 16B chunks; 8 waves -> 2 DMA instr each.
  auto stage = [&](int buf, int k0) {
#pragma unroll
    for (int j = 0; j < 2; ++j) {
      int c   = wid * 64 + j * 32 + lane;  // 0..511
      int row = c >> 2;
      int col = (c & 3) * 8;
      async_copy_b128(lds_off(&ldsA[buf][row][col]),
                      &A[(size_t)(bm + row) * K + k0 + col]);
      async_copy_b128(lds_off(&ldsW[buf][row][col]),
                      &WT[(size_t)(bn + row) * K + k0 + col]);
    }
  };

  stage(0, 0);
  const int nk = K / 32;
  for (int i = 0; i < nk; ++i) {
    const int buf = i & 1;
    wait_async();
    __syncthreads();
    if (i + 1 < nk) stage(buf ^ 1, (i + 1) * 32);

    bf16x16 af[2], bfr[4];
#pragma unroll
    for (int mt = 0; mt < 2; ++mt) {
      int r  = wm + mt * 16 + laneR;
      int ko = laneH * 16;  // A layout: lanes0-15 K0-15; lanes16-31 K16-31
      af[mt] = ld_frag(&ldsA[buf][r][ko], &ldsA[buf][r][ko + 8]);
    }
#pragma unroll
    for (int nt = 0; nt < 4; ++nt) {
      int n  = wn + nt * 16 + laneR;
      int ko = laneH * 16;  // B layout: lanes0-15 K0-15; lanes16-31 K16-31
      bfr[nt] = ld_frag(&ldsW[buf][n][ko], &ldsW[buf][n][ko + 8]);
    }
#pragma unroll
    for (int mt = 0; mt < 2; ++mt)
#pragma unroll
      for (int nt = 0; nt < 4; ++nt)
        acc[mt][nt] = wmma_bf16(af[mt], bfr[nt], acc[mt][nt]);
  }

  // C layout: VGPR j -> row j (lanes 0-15) / j+8 (lanes 16-31)
#pragma unroll
  for (int mt = 0; mt < 2; ++mt) {
#pragma unroll
    for (int nt = 0; nt < 4; ++nt) {
      int n = bn + wn + nt * 16 + laneR;
      float bv = bias ? bias[n] : 0.0f;
#pragma unroll
      for (int j = 0; j < 8; ++j) {
        int m = bm + wm + mt * 16 + laneH * 8 + j;
        float v = acc[mt][nt][j] + bv;
        if constexpr (OUT_F32) ((float*)Cout)[(size_t)m * N + n] = v;
        else                   ((bf16_t*)Cout)[(size_t)m * N + n] = (bf16_t)v;
      }
    }
  }
}

// ---------------------------------------------------------------------------
// Flash attention with Transformer-XL relative-position bias.
// 128 threads (4 waves), 64 query rows of one (b,h) per block.
// K and P tiles are straight row-major copies -> async DMA to LDS; V needs a
// transpose so it stays on the manual path. Scores via chained WMMAs,
// online softmax in fp32, O += softmax(P) @ V via WMMA.
// ---------------------------------------------------------------------------
__global__ __launch_bounds__(128) void attn_kernel(
    const bf16_t* __restrict__ Q,   // (B,T,NF)
    const bf16_t* __restrict__ Kk,  // (B,T,NF)
    const bf16_t* __restrict__ V,   // (B,T,NF)
    const bf16_t* __restrict__ P,   // (T,NF)
    const float*  __restrict__ mask,    // (B,1,T), indexed by key position
    const float*  __restrict__ bias_u,  // (H,64)
    const float*  __restrict__ bias_v,  // (H,64)
    bf16_t* __restrict__ Out)           // (B,T,NF)
{
  __shared__ bf16_t ldsK [64][64];   // (s, d)
  __shared__ bf16_t ldsVT[64][64];   // (d, s)
  __shared__ bf16_t ldsP [64][64];   // (s, d)
  __shared__ bf16_t ldsQU[64][64];   // (t, d) q + pos_bias_u
  __shared__ bf16_t ldsQV[64][64];   // (t, d) q + pos_bias_v
  __shared__ bf16_t ldsS [4][16][64];// per-wave softmax tile (C->A relayout)

  const int tid   = threadIdx.x;
  const int lane  = tid & 31;
  const int wid   = tid >> 5;
  const int laneR = lane & 15;
  const int laneH = lane >> 4;

  const int qBase = blockIdx.x * 64;
  const int bh    = blockIdx.y;
  const int b     = bh >> 3;
  const int h     = bh & 7;

  // ---- stage qu / qv tiles (fp32 bias add, bf16 store) ----
  {
    int r  = tid >> 1;
    int c0 = (tid & 1) * 32;
    const bf16_t* qrow = &Q[(size_t)(b * NT + qBase + r) * NF + h * ND];
#pragma unroll
    for (int i = 0; i < 32; ++i) {
      float q = (float)qrow[c0 + i];
      ldsQU[r][c0 + i] = (bf16_t)(q + bias_u[h * ND + c0 + i]);
      ldsQV[r][c0 + i] = (bf16_t)(q + bias_v[h * ND + c0 + i]);
    }
  }
  __syncthreads();

  bf16x16 au[2], av[2];
#pragma unroll
  for (int kc = 0; kc < 2; ++kc) {
    int r  = wid * 16 + laneR;
    int ko = kc * 32 + laneH * 16;
    au[kc] = ld_frag(&ldsQU[r][ko], &ldsQU[r][ko + 8]);
    av[kc] = ld_frag(&ldsQV[r][ko], &ldsQV[r][ko + 8]);
  }

  f32x8 oacc[4];
#pragma unroll
  for (int nt = 0; nt < 4; ++nt) oacc[nt] = zero8();
  float mrun[8], lrun[8];
#pragma unroll
  for (int j = 0; j < 8; ++j) { mrun[j] = -1e30f; lrun[j] = 0.0f; }

  for (int s0 = 0; s0 < NT; s0 += 64) {
    __syncthreads();  // previous iteration done reading K/VT/P
    {
      // K, P: async DMA (64x64 bf16 = 512 chunks of 16B; 128 lanes x 4)
#pragma unroll
      for (int j = 0; j < 4; ++j) {
        int c   = j * 128 + tid;
        int r   = c >> 3;
        int col = (c & 7) * 8;
        async_copy_b128(lds_off(&ldsK[r][col]),
                        &Kk[(size_t)(b * NT + s0 + r) * NF + h * ND + col]);
        async_copy_b128(lds_off(&ldsP[r][col]),
                        &P[(size_t)(s0 + r) * NF + h * ND + col]);
      }
      // V: manual transposed stage
      int r  = tid >> 1;
      int c0 = (tid & 1) * 32;
      const bf16_t* vs = &V[(size_t)(b * NT + s0 + r) * NF + h * ND + c0];
#pragma unroll
      for (int i = 0; i < 32; ++i) ldsVT[c0 + i][r] = vs[i];
      wait_async();
    }
    __syncthreads();

    // ---- scores: (q+u)@K^T + (q+v)@P^T ----
    f32x8 sc[4];
#pragma unroll
    for (int nt = 0; nt < 4; ++nt) {
      f32x8 c = zero8();
      int n = nt * 16 + laneR;
#pragma unroll
      for (int kc = 0; kc < 2; ++kc) {
        int ko = kc * 32 + laneH * 16;
        bf16x16 bk = ld_frag(&ldsK[n][ko], &ldsK[n][ko + 8]);
        c = wmma_bf16(au[kc], bk, c);
        bf16x16 bp = ld_frag(&ldsP[n][ko], &ldsP[n][ko + 8]);
        c = wmma_bf16(av[kc], bp, c);
      }
      float mv = mask[(size_t)b * NT + s0 + nt * 16 + laneR];
#pragma unroll
      for (int j = 0; j < 8; ++j) c[j] = c[j] * 0.125f + mv;  // 1/sqrt(64)
      sc[nt] = c;
    }

    // ---- online softmax (row j lives in accumulator slot j per half) ----
    float mnew[8];
#pragma unroll
    for (int j = 0; j < 8; ++j) {
      float v = fmaxf(fmaxf(sc[0][j], sc[1][j]), fmaxf(sc[2][j], sc[3][j]));
#pragma unroll
      for (int off = 1; off < 16; off <<= 1)
        v = fmaxf(v, __shfl_xor(v, off, 32));
      mnew[j] = fmaxf(mrun[j], v);
    }
#pragma unroll
    for (int j = 0; j < 8; ++j) {
      float alpha = __expf(mrun[j] - mnew[j]);
      lrun[j] *= alpha;
#pragma unroll
      for (int nt = 0; nt < 4; ++nt) oacc[nt][j] *= alpha;
      mrun[j] = mnew[j];
    }
#pragma unroll
    for (int nt = 0; nt < 4; ++nt) {
#pragma unroll
      for (int j = 0; j < 8; ++j) {
        float p = __expf(sc[nt][j] - mrun[j]);
        sc[nt][j] = p;
        ldsS[wid][j + laneH * 8][nt * 16 + laneR] = (bf16_t)p;  // C -> row-major
      }
    }
#pragma unroll
    for (int j = 0; j < 8; ++j) {
      float v = sc[0][j] + sc[1][j] + sc[2][j] + sc[3][j];
#pragma unroll
      for (int off = 1; off < 16; off <<= 1) v += __shfl_xor(v, off, 32);
      lrun[j] += v;
    }

    // ---- O += softmax(P) @ V ----
#pragma unroll
    for (int kc = 0; kc < 2; ++kc) {
      int ko = kc * 32 + laneH * 16;
      bf16x16 pa = ld_frag(&ldsS[wid][laneR][ko], &ldsS[wid][laneR][ko + 8]);
#pragma unroll
      for (int nt = 0; nt < 4; ++nt) {
        int d   = nt * 16 + laneR;
        int ko2 = kc * 32 + laneH * 16;
        bf16x16 bv = ld_frag(&ldsVT[d][ko2], &ldsVT[d][ko2 + 8]);
        oacc[nt] = wmma_bf16(pa, bv, oacc[nt]);
      }
    }
  }

  // ---- finalize: O /= l, store bf16 (b, t, h, d) ----
#pragma unroll
  for (int nt = 0; nt < 4; ++nt) {
    int d = nt * 16 + laneR;
#pragma unroll
    for (int j = 0; j < 8; ++j) {
      int t = qBase + wid * 16 + j + laneH * 8;
      float o = oacc[nt][j] / lrun[j];
      Out[(size_t)(b * NT + t) * NF + h * ND + d] = (bf16_t)o;
    }
  }
}

// ---------------------------------------------------------------------------
// Launch: convert + transpose prep, 4 projection GEMMs, attention, out GEMM.
// ---------------------------------------------------------------------------
extern "C" void kernel_launch(void* const* d_in, const int* in_sizes, int n_in,
                              void* d_out, int out_size, void* d_ws, size_t ws_size,
                              hipStream_t stream) {
  const float* query   = (const float*)d_in[0];
  const float* key     = (const float*)d_in[1];
  const float* value   = (const float*)d_in[2];
  const float* pos_emb = (const float*)d_in[3];
  const float* mask    = (const float*)d_in[4];
  const float* Wq = (const float*)d_in[5];
  const float* bq = (const float*)d_in[6];
  const float* Wk = (const float*)d_in[7];
  const float* bk = (const float*)d_in[8];
  const float* Wv = (const float*)d_in[9];
  const float* bv = (const float*)d_in[10];
  const float* Wo = (const float*)d_in[11];
  const float* bo = (const float*)d_in[12];
  const float* Wpos = (const float*)d_in[13];
  const float* pbu  = (const float*)d_in[14];
  const float* pbv  = (const float*)d_in[15];

  const int M = NB * NT;  // 16384

  char* ws = (char*)d_ws;
  size_t off = 0;
  auto alloc = [&](size_t bytes) -> char* {
    char* p = ws + off;
    off += (bytes + 255) & ~(size_t)255;
    return p;
  };
  bf16_t* xq  = (bf16_t*)alloc((size_t)M * NF * 2);
  bf16_t* xk  = (bf16_t*)alloc((size_t)M * NF * 2);
  bf16_t* xv  = (bf16_t*)alloc((size_t)M * NF * 2);
  bf16_t* xp  = (bf16_t*)alloc((size_t)NT * NF * 2);
  bf16_t* qb  = (bf16_t*)alloc((size_t)M * NF * 2);
  bf16_t* kb  = (bf16_t*)alloc((size_t)M * NF * 2);
  bf16_t* vb  = (bf16_t*)alloc((size_t)M * NF * 2);
  bf16_t* pb  = (bf16_t*)alloc((size_t)NT * NF * 2);
  bf16_t* wqT = (bf16_t*)alloc((size_t)NF * NF * 2);
  bf16_t* wkT = (bf16_t*)alloc((size_t)NF * NF * 2);
  bf16_t* wvT = (bf16_t*)alloc((size_t)NF * NF * 2);
  bf16_t* woT = (bf16_t*)alloc((size_t)NF * NF * 2);
  bf16_t* wpT = (bf16_t*)alloc((size_t)NF * NF * 2);
  bf16_t* attn = xq;  // xq is dead after the Q projection; reuse (deterministic)

  const int nAct = M * NF;       // 8388608
  const int nPos = NT * NF;      // 524288
  dim3 cvtG(nAct / (256 * 8));   // 4096
  dim3 cvtP(nPos / (256 * 8));   // 256
  cvt_bf16<<<cvtG, 256, 0, stream>>>(query,   xq, nAct);
  cvt_bf16<<<cvtG, 256, 0, stream>>>(key,     xk, nAct);
  cvt_bf16<<<cvtG, 256, 0, stream>>>(value,   xv, nAct);
  cvt_bf16<<<cvtP, 256, 0, stream>>>(pos_emb, xp, nPos);

  dim3 trG(NF / 32, NF / 32);    // (16,16)
  wtrans_bf16<<<trG, 256, 0, stream>>>(Wq,   wqT, NF, NF);
  wtrans_bf16<<<trG, 256, 0, stream>>>(Wk,   wkT, NF, NF);
  wtrans_bf16<<<trG, 256, 0, stream>>>(Wv,   wvT, NF, NF);
  wtrans_bf16<<<trG, 256, 0, stream>>>(Wo,   woT, NF, NF);
  wtrans_bf16<<<trG, 256, 0, stream>>>(Wpos, wpT, NF, NF);

  dim3 gProj(NF / 128, M / 128);   // (4, 128)
  dim3 gPos(NF / 128, NT / 128);   // (4, 8)
  gemm_bf16tt<false><<<gProj, 256, 0, stream>>>(xq, wqT, bq, qb, M, NF, NF);
  gemm_bf16tt<false><<<gProj, 256, 0, stream>>>(xk, wkT, bk, kb, M, NF, NF);
  gemm_bf16tt<false><<<gProj, 256, 0, stream>>>(xv, wvT, bv, vb, M, NF, NF);
  gemm_bf16tt<false><<<gPos,  256, 0, stream>>>(xp, wpT, (const float*)nullptr,
                                                pb, NT, NF, NF);

  attn_kernel<<<dim3(NT / 64, NB * NH), 128, 0, stream>>>(qb, kb, vb, pb, mask,
                                                          pbu, pbv, attn);

  gemm_bf16tt<true><<<gProj, 256, 0, stream>>>(attn, woT, bo, (float*)d_out,
                                               M, NF, NF);
}
